// PiCANet_L_39968965656962
// MI455X (gfx1250) — matrix operand order repaired and checked by reading, not compiled, not run on hardware
//
#include <hip/hip_runtime.h>

// ---------------------------------------------------------------------------
// PiCANet-L fused pipeline for MI455X (gfx1250, wave32, WMMA bf16)
//   B=16, C=64, H=W=56, OC1=128, taps=49 (7x7 dil=2, pad=6), OC2=49
// K ordering for conv1 GEMM: k = tap*64 + ic  (ic-minor -> contiguous frags)
// ---------------------------------------------------------------------------

#define BB   16
#define CIN  64
#define HH   56
#define WW   56
#define HWP  (HH * WW)          // 3136
#define OC1  128
#define KTOT (49 * CIN)         // 3136
#define NPIX (BB * HWP)         // 50176
#define ICP  72                 // padded ic stride in LDS (bank-conflict break)

typedef __attribute__((ext_vector_type(16))) __bf16          v16bf;
typedef __attribute__((ext_vector_type(8)))  __bf16          v8bf;
typedef __attribute__((ext_vector_type(8)))  float           v8f;
typedef __attribute__((ext_vector_type(8)))  unsigned short  us8;

__device__ __forceinline__ unsigned short f32_to_bf16_bits(float f) {
    unsigned int u = __float_as_uint(f);
    u += 0x7FFFu + ((u >> 16) & 1u);      // round-to-nearest-even
    return (unsigned short)(u >> 16);
}

__device__ __forceinline__ v16bf join8(us8 lo, us8 hi) {
    v8bf a = __builtin_bit_cast(v8bf, lo);
    v8bf b = __builtin_bit_cast(v8bf, hi);
    return __builtin_shufflevector(a, b, 0, 1, 2, 3, 4, 5, 6, 7,
                                          8, 9, 10, 11, 12, 13, 14, 15);
}

// ---------------------------------------------------------------------------
// weight conversion / permutation kernels
// ---------------------------------------------------------------------------
__global__ void cvt_w1(const float* __restrict__ w1, unsigned short* __restrict__ w1k) {
    int i = blockIdx.x * 256 + threadIdx.x;          // dst: [128][tap*64+ic]
    if (i < OC1 * KTOT) {
        int oc  = i / KTOT;
        int r   = i - oc * KTOT;
        int tap = r >> 6;
        int ic  = r & 63;
        w1k[i] = f32_to_bf16_bits(w1[(oc * CIN + ic) * 49 + tap]);
    }
}

__global__ void cvt_w2(const float* __restrict__ w2, unsigned short* __restrict__ w2b) {
    int i = blockIdx.x * 256 + threadIdx.x;          // padded [64][128]
    if (i < 64 * 128) {
        int oc = i >> 7, k = i & 127;
        float v = (oc < 49) ? w2[oc * 128 + k] : 0.0f;
        w2b[i] = f32_to_bf16_bits(v);
    }
}

// ---------------------------------------------------------------------------
// conv1: 7x7 dilation-2 implicit GEMM, bf16 WMMA, f32 accumulate.
// grid = B*H*4 blocks; block = 256 threads = 8 waves; wave w -> oc tile w*16.
// LDS patch xs[p][col][ic] (ic-contiguous) so B fragments are 2x ds_load_b128.
// Output stored TRANSPOSED: y1t[b][pixel][oc] (bf16) for conv2's fragments.
// ---------------------------------------------------------------------------
__global__ __launch_bounds__(256)
void conv1_wmma(const float* __restrict__ x,
                const unsigned short* __restrict__ w1k,   // bf16 [128][3136]
                const float* __restrict__ b1,
                unsigned short* __restrict__ y1t)         // bf16 [B][3136][128]
{
    const int blk = blockIdx.x;
    const int wt  = blk & 3;
    const int h   = (blk >> 2) % HH;
    const int b   = (blk >> 2) / HH;
    const int w0  = wt * 16;

    __shared__ __align__(32) unsigned short xs[7][28][ICP];   // 28224 B

    const int tid = threadIdx.x;
    // cooperative stage: 12544 elems / 256 thr = 49 each; coalesced x reads
    for (int i = tid; i < 7 * 28 * CIN; i += 256) {
        int col = i % 28;
        int p   = (i / 28) % 7;
        int ic  = i / 196;
        int xr  = h + 2 * p - 6;
        int xc  = w0 - 6 + col;
        float v = 0.0f;
        if (xr >= 0 && xr < HH && xc >= 0 && xc < WW)
            v = x[((b * CIN + ic) * HH + xr) * WW + xc];
        xs[p][col][ic] = f32_to_bf16_bits(v);
    }
    __syncthreads();

    const int wave = tid >> 5;
    const int lane = tid & 31;
    const int row  = lane & 15;      // A: M row (oc), B: N col (pixel), C: N col
    const int hi   = lane >> 4;      // lane half selects K sub-block
    const int oc0  = wave * 16;

    v8f acc = {};
    const unsigned short* arow = w1k + (oc0 + row) * KTOT;

    #pragma unroll 2
    for (int kbase = 0; kbase < KTOT; kbase += 32) {
        // uniform across the wave: one tap per 32-wide K chunk (64 ic per tap)
        int tap = kbase >> 6;
        int p   = tap / 7;
        int q   = tap - p * 7;
        int icb = (kbase & 32) + hi * 8;

        // ---- A fragment: w1k row, k = kbase + {hi*8+e, 16+hi*8+e}
        const unsigned short* ap = arow + kbase + hi * 8;
        __builtin_prefetch(ap + 128, 0, 1);           // global_prefetch next chunks
        us8 alo = *(const us8*)ap;
        us8 ahi = *(const us8*)(ap + 16);

        // ---- B fragment: pixel col = row, ic = icb+{e, 16+e} at fixed tap
        const unsigned short* bp = &xs[p][row + 2 * q][icb];
        us8 blo = *(const us8*)bp;
        us8 bhi = *(const us8*)(bp + 16);

        acc = __builtin_amdgcn_wmma_f32_16x16x32_bf16(
                  false, join8(alo, ahi), false, join8(blo, bhi),
                  (short)0, acc, false, false);
    }

    // ---- epilogue: C layout VGPR r -> M = hi*8 + r (oc), N = row (pixel)
    const int w_ = w0 + row;
    if (w_ < WW) {
        const int pix = h * WW + w_;
        us8 o;
        #pragma unroll
        for (int r = 0; r < 8; ++r)
            o[r] = f32_to_bf16_bits(acc[r] + b1[oc0 + hi * 8 + r]);
        *(us8*)&y1t[(size_t)(b * HWP + pix) * OC1 + oc0 + hi * 8] = o;
    }
}

// ---------------------------------------------------------------------------
// conv2 (1x1, 128->49 padded to 64) + softmax over 49 channels.
// grid = NPIX/16 blocks; block = 1 wave (32 threads). 16 WMMAs per tile.
// y1t is pixel-major so each B fragment is two contiguous 16-byte loads per lane.
// ---------------------------------------------------------------------------
__global__ __launch_bounds__(32)
void conv2_softmax(const unsigned short* __restrict__ y1t,  // bf16 [B][3136][128]
                   const unsigned short* __restrict__ w2b,  // bf16 [64][128]
                   const float* __restrict__ b2,
                   float* __restrict__ attn)                // f32 [B][49][3136]
{
    const int tile = blockIdx.x;               // 3136 tiles of 16 pixels
    const int b    = tile / (HWP / 16);
    const int ps0  = (tile % (HWP / 16)) * 16;
    const int lane = threadIdx.x;
    const int row  = lane & 15;
    const int hi   = lane >> 4;

    __shared__ float logits[64][16];

    // B fragments shared across the 4 M-tiles: build once (contiguous loads)
    const unsigned short* yb = y1t + (size_t)(b * HWP + ps0 + row) * OC1;
    v16bf Bfr[4];
    #pragma unroll
    for (int kc = 0; kc < 4; ++kc) {
        const unsigned short* bp = yb + kc * 32 + hi * 8;
        us8 blo = *(const us8*)bp;
        us8 bhi = *(const us8*)(bp + 16);
        Bfr[kc] = join8(blo, bhi);
    }

    #pragma unroll
    for (int mt = 0; mt < 4; ++mt) {
        v8f acc = {};
        const unsigned short* arow = w2b + (mt * 16 + row) * 128;
        #pragma unroll
        for (int kc = 0; kc < 4; ++kc) {
            const unsigned short* ap = arow + kc * 32 + hi * 8;
            us8 alo = *(const us8*)ap;
            us8 ahi = *(const us8*)(ap + 16);
            acc = __builtin_amdgcn_wmma_f32_16x16x32_bf16(
                      false, join8(alo, ahi), false, Bfr[kc],
                      (short)0, acc, false, false);
        }
        #pragma unroll
        for (int r = 0; r < 8; ++r)
            logits[mt * 16 + hi * 8 + r][row] = acc[r];
    }
    __syncthreads();

    // softmax over 49 channels, one pixel per thread (lanes 0..15)
    if (lane < 16) {
        float m = -3.4e38f;
        #pragma unroll 1
        for (int k = 0; k < 49; ++k)
            m = fmaxf(m, logits[k][lane] + b2[k]);
        float s = 0.0f;
        #pragma unroll 1
        for (int k = 0; k < 49; ++k)
            s += __expf(logits[k][lane] + b2[k] - m);
        float inv = 1.0f / s;
        #pragma unroll 1
        for (int k = 0; k < 49; ++k)
            attn[(b * 49 + k) * HWP + ps0 + lane] =
                __expf(logits[k][lane] + b2[k] - m) * inv;
    }
}

// ---------------------------------------------------------------------------
// final: out[b,c,h,w] = sum_k x[b,c,h+2p-6,w+2q-6] * attn_flat[b*153664 +
//        c*2401 + (s/64)*49 + k]   (torch .view row-major correspondence)
// ---------------------------------------------------------------------------
__global__ __launch_bounds__(256)
void final_mix(const float* __restrict__ x,
               const float* __restrict__ attn,
               float* __restrict__ out)
{
    int idx = blockIdx.x * 256 + threadIdx.x;
    if (idx >= BB * CIN * HWP) return;
    int w = idx % WW;
    int h = (idx / WW) % HH;
    int c = (idx / HWP) % CIN;
    int b = idx / (CIN * HWP);
    int s = h * WW + w;

    const float* ap = attn + b * (49 * HWP) + c * 2401 + (s >> 6) * 49;
    const float* xp = x + (b * CIN + c) * HWP;

    float sum = 0.0f;
    #pragma unroll
    for (int p = 0; p < 7; ++p) {
        int xr = h + 2 * p - 6;
        bool rok = (xr >= 0) && (xr < HH);
        #pragma unroll
        for (int q = 0; q < 7; ++q) {
            int xc = w + 2 * q - 6;
            float xv = (rok && xc >= 0 && xc < WW) ? xp[xr * WW + xc] : 0.0f;
            sum = fmaf(xv, ap[p * 7 + q], sum);
        }
    }
    out[idx] = sum;
}

// ---------------------------------------------------------------------------
// launcher
// ---------------------------------------------------------------------------
extern "C" void kernel_launch(void* const* d_in, const int* in_sizes, int n_in,
                              void* d_out, int out_size, void* d_ws, size_t ws_size,
                              hipStream_t stream) {
    const float* x  = (const float*)d_in[0];   // [16,64,56,56]
    const float* w1 = (const float*)d_in[1];   // [128,64,7,7]
    const float* b1 = (const float*)d_in[2];   // [128]
    const float* w2 = (const float*)d_in[3];   // [49,128,1,1]
    const float* b2 = (const float*)d_in[4];   // [49]
    float* out = (float*)d_out;

    // workspace carve-up (all 256B aligned)
    char* ws = (char*)d_ws;
    size_t off = 0;
    auto carve = [&](size_t bytes) { char* p = ws + off;
                                     off += (bytes + 255) & ~(size_t)255; return p; };
    unsigned short* w1k = (unsigned short*)carve((size_t)OC1 * KTOT * 2);     // 0.80 MB
    unsigned short* w2b = (unsigned short*)carve((size_t)64 * 128 * 2);       // 16 KB
    unsigned short* y1t = (unsigned short*)carve((size_t)BB * HWP * OC1 * 2); // 12.8 MB
    float*          att = (float*)carve((size_t)BB * 49 * HWP * 4);           // 9.8 MB
    (void)ws_size; (void)in_sizes; (void)n_in; (void)out_size;

    cvt_w1<<<(OC1 * KTOT + 255) / 256, 256, 0, stream>>>(w1, w1k);
    cvt_w2<<<(64 * 128 + 255) / 256, 256, 0, stream>>>(w2, w2b);

    conv1_wmma<<<BB * HH * 4, 256, 0, stream>>>(x, w1k, b1, y1t);

    conv2_softmax<<<NPIX / 16, 32, 0, stream>>>(y1t, w2b, b2, att);

    final_mix<<<(BB * CIN * HWP + 255) / 256, 256, 0, stream>>>(x, att, out);
}
